// mSWEGNNLayer_69827578298480
// MI455X (gfx1250) — compile-verified, matrix operand order checked by reading, not run on hardware
//
#include <hip/hip_runtime.h>
#include <hip/hip_bf16.h>

typedef __attribute__((ext_vector_type(16))) __bf16  v16bf;
typedef __attribute__((ext_vector_type(8)))  float   v8f;
typedef __attribute__((ext_vector_type(4)))  float   v4f;
typedef __attribute__((ext_vector_type(4)))  unsigned int v4u;

#define DD   64
#define HH   128
#define PSI  320

union BF16Frag {
    v16bf v;
    unsigned short u[16];
    v4u q[2];
};

// Gathered fp32 source for one 32-wide K chunk of an A fragment
struct F4x4 { v4f f0, f1, f2, f3; };

// base is a wave-uniform pointer (SGPR), elemOff a 32-bit lane offset -> saddr loads
__device__ __forceinline__ F4x4 loadChunk(const float* base, int elemOff) {
    const v4f* q4 = (const v4f*)(base + elemOff);
    F4x4 r;
    r.f0 = q4[0];   // K = base + 0..3
    r.f1 = q4[1];   // K = base + 4..7
    r.f2 = q4[4];   // K = base + 16..19
    r.f3 = q4[5];   // K = base + 20..23
    return r;
}

__device__ __forceinline__ void cvtFrag(BF16Frag& a, const F4x4& f) {
    #pragma unroll
    for (int j = 0; j < 4; ++j) {
        a.v[j]      = (__bf16)f.f0[j];
        a.v[4 + j]  = (__bf16)f.f1[j];
        a.v[8 + j]  = (__bf16)f.f2[j];
        a.v[12 + j] = (__bf16)f.f3[j];
    }
}

__device__ __forceinline__ void loadB(BF16Frag& bf, const unsigned short* base,
                                      int slot, int lane) {
    const v4u* bp = (const v4u*)(base + ((slot * 32 + lane) << 4));
    bf.q[0] = bp[0];
    bf.q[1] = bp[1];
}

__global__ void zero_kernel(float* __restrict__ p, long n) {
    long i = (long)blockIdx.x * blockDim.x + threadIdx.x;
    long stride = (long)gridDim.x * blockDim.x;
    for (; i < n; i += stride) p[i] = 0.0f;
}

// ---------------------------------------------------------------------------
// Fragment index math (CDNA5 16-bit WMMA layouts, wave32):
//   A 16x32:  lane L (m = L&15, h = L>>4) holds K = (i&8)*2 + h*8 + (i&7):
//   two contiguous 8-elem runs [h*8, h*8+8) and [16+h*8, +8).
//   B 32x16 mirrors this with N = L&15.
// Weight staging inverse: kk = K%32: h = (kk>>3)&1, i = ((kk&16)>>1)|(kk&7).
//
// LDS layout (dynamic, ~194 KB):
//   [0,128KB):        weight B-fragments (halfwords):
//       w0: base 0     count 10*8*32*16 = 40960  (slot = c*8+t)
//       w1: base 40960 count  4*8*32*16 = 16384  (slot = c*8+t)
//       w2: base 57344 count  4*4*32*16 =  8192  (slot = c*4+t)
//   [128KB,192KB):    per-wave activation staging: 2 tiles x 16x128 bf16 = 8KB/wave
//   [192KB,192KB+1280): biases b0|b1|b2 as fp32
// Each wave processes a PAIR of 16-edge tiles: every B fragment read from LDS
// feeds two WMMAs -> halves LDS B-traffic per edge.
// ---------------------------------------------------------------------------
__global__ void __launch_bounds__(256)
edge_mlp_kernel(const float* __restrict__ h_d, const float* __restrict__ h_s,
                const float* __restrict__ edge_feat,
                const int* __restrict__ sender, const int* __restrict__ receiver,
                const float* __restrict__ W0, const float* __restrict__ b0,
                const float* __restrict__ W1, const float* __restrict__ b1,
                const float* __restrict__ W2, const float* __restrict__ b2,
                float* __restrict__ agg, int numPairs)
{
    extern __shared__ char smem[];
    unsigned short* wfrag  = (unsigned short*)smem;
    unsigned short* w1base = wfrag + 40960;
    unsigned short* w2base = wfrag + 57344;
    __bf16* xstage = (__bf16*)(smem + 131072);
    float*  biasL  = (float*)(smem + 196608);

    const int tid = threadIdx.x;

    // ---- stage MLP weights (B-fragment layout) + biases into LDS ----
    for (int idx = tid; idx < 65536; idx += 256) {
        int k, n, T, base; float val;
        if (idx < 40960)       { k = idx >> 7;            n = idx & 127; T = 8; base = 0;     val = W0[idx]; }
        else if (idx < 57344)  { int j = idx - 40960; k = j >> 7; n = j & 127; T = 8; base = 40960; val = W1[j]; }
        else                   { int j = idx - 57344; k = j >> 6; n = j & 63;  T = 4; base = 57344; val = W2[j]; }
        int c  = k >> 5, kk = k & 31;
        int hh = (kk >> 3) & 1;
        int ii = ((kk & 16) >> 1) | (kk & 7);
        int t  = n >> 4;
        int ln = (n & 15) + (hh << 4);
        wfrag[base + (((c * T + t) * 32 + ln) << 4) + ii] = __builtin_bit_cast(unsigned short, (__bf16)val);
    }
    if (tid < 320) {
        float v = (tid < 128) ? b0[tid] : (tid < 256 ? b1[tid - 128] : b2[tid - 256]);
        biasL[tid] = v;
    }
    __syncthreads();

    const int lane = tid & 31;
    const int wave = tid >> 5;
    const int nlo  = lane & 15;
    const int h    = lane >> 4;
    __bf16* xsA = xstage + wave * 4096;     // 16x128 bf16
    __bf16* xsB = xsA + 2048;

    for (long pair = (long)blockIdx.x * 8 + wave; pair < numPairs; pair += (long)gridDim.x * 8) {
        const int baseA = (int)pair * 32;
        const int baseB = baseA + 16;
        const int eA = baseA + nlo, eB = baseB + nlo;
        // 32-bit element offsets into the three gather tables (saddr-form loads)
        const int oSA = sender[eA] * DD, oRA = receiver[eA] * DD, oEA = eA * DD;
        const int oSB = sender[eB] * DD, oRB = receiver[eB] * DD, oEB = eB * DD;

        // ================= layer 0: [16,320]@[320,128], x2 tiles =================
        v8f accA[8], accB[8];
        #pragma unroll
        for (int t = 0; t < 8; ++t) {
            float bb = biasL[t * 16 + nlo];
            #pragma unroll
            for (int j = 0; j < 8; ++j) { accA[t][j] = bb; accB[t][j] = bb; }
        }

        F4x4 cfA = loadChunk(h_s, oSA + h * 8);
        F4x4 cfB = loadChunk(h_s, oSB + h * 8);
        BF16Frag bfb[2];
        loadB(bfb[0], wfrag, 0, lane);

        #pragma unroll
        for (int c = 0; c < 10; ++c) {
            BF16Frag aA, aB;
            cvtFrag(aA, cfA);
            cvtFrag(aB, cfB);
            if (c < 9) {
                const int cn  = c + 1;
                const int sg  = cn >> 1;
                const int sub = (cn & 1) * 32 + h * 8;
                const float* gb = (sg < 2) ? h_s : (sg < 4 ? h_d : edge_feat);
                const int offA0 = (sg == 0 || sg == 2) ? oSA : ((sg == 1 || sg == 3) ? oRA : oEA);
                const int offB0 = (sg == 0 || sg == 2) ? oSB : ((sg == 1 || sg == 3) ? oRB : oEB);
                cfA = loadChunk(gb, offA0 + sub);
                cfB = loadChunk(gb, offB0 + sub);
            }
            #pragma unroll
            for (int t = 0; t < 8; ++t) {
                const int step = c * 8 + t;            // == LDS slot
                if (step < 79) loadB(bfb[(step + 1) & 1], wfrag, step + 1, lane);
                accA[t] = __builtin_amdgcn_wmma_f32_16x16x32_bf16(
                              false, aA.v, false, bfb[step & 1].v, (short)0, accA[t], false, false);
                accB[t] = __builtin_amdgcn_wmma_f32_16x16x32_bf16(
                              false, aB.v, false, bfb[step & 1].v, (short)0, accB[t], false, false);
            }
        }
        #pragma unroll
        for (int t = 0; t < 8; ++t)
            #pragma unroll
            for (int r = 0; r < 8; ++r) {
                xsA[(r + h * 8) * 128 + t * 16 + nlo] = (__bf16)fmaxf(accA[t][r], 0.0f);
                xsB[(r + h * 8) * 128 + t * 16 + nlo] = (__bf16)fmaxf(accB[t][r], 0.0f);
            }
        asm volatile("s_wait_dscnt 0" ::: "memory");

        // ================= layer 1: [16,128]@[128,128], x2 tiles =================
        BF16Frag a1A[4], a1B[4];
        #pragma unroll
        for (int c = 0; c < 4; ++c) {
            a1A[c].q[0] = *(const v4u*)(xsA + nlo * 128 + c * 32 + h * 8);
            a1A[c].q[1] = *(const v4u*)(xsA + nlo * 128 + c * 32 + 16 + h * 8);
            a1B[c].q[0] = *(const v4u*)(xsB + nlo * 128 + c * 32 + h * 8);
            a1B[c].q[1] = *(const v4u*)(xsB + nlo * 128 + c * 32 + 16 + h * 8);
        }
        loadB(bfb[0], w1base, 0, lane);
        #pragma unroll
        for (int t = 0; t < 8; ++t) {
            v8f accA1, accB1;
            float bb = biasL[128 + t * 16 + nlo];
            #pragma unroll
            for (int j = 0; j < 8; ++j) { accA1[j] = bb; accB1[j] = bb; }
            #pragma unroll
            for (int c = 0; c < 4; ++c) {
                const int step = t * 4 + c;
                if (step < 31) {
                    const int tn = (c < 3) ? t : t + 1;
                    const int cn = (c < 3) ? c + 1 : 0;
                    loadB(bfb[(step + 1) & 1], w1base, cn * 8 + tn, lane);
                }
                accA1 = __builtin_amdgcn_wmma_f32_16x16x32_bf16(
                            false, a1A[c].v, false, bfb[step & 1].v, (short)0, accA1, false, false);
                accB1 = __builtin_amdgcn_wmma_f32_16x16x32_bf16(
                            false, a1B[c].v, false, bfb[step & 1].v, (short)0, accB1, false, false);
            }
            #pragma unroll
            for (int r = 0; r < 8; ++r) {
                xsA[(r + h * 8) * 128 + t * 16 + nlo] = (__bf16)fmaxf(accA1[r], 0.0f);
                xsB[(r + h * 8) * 128 + t * 16 + nlo] = (__bf16)fmaxf(accB1[r], 0.0f);
            }
        }
        asm volatile("s_wait_dscnt 0" ::: "memory");

        // ================= layer 2 + s_ij + scatter-add, x2 tiles =================
        BF16Frag a2A[4], a2B[4];
        #pragma unroll
        for (int c = 0; c < 4; ++c) {
            a2A[c].q[0] = *(const v4u*)(xsA + nlo * 128 + c * 32 + h * 8);
            a2A[c].q[1] = *(const v4u*)(xsA + nlo * 128 + c * 32 + 16 + h * 8);
            a2B[c].q[0] = *(const v4u*)(xsB + nlo * 128 + c * 32 + h * 8);
            a2B[c].q[1] = *(const v4u*)(xsB + nlo * 128 + c * 32 + 16 + h * 8);
        }
        int smA[8], rmA[8], smB[8], rmB[8];
        #pragma unroll
        for (int r = 0; r < 8; ++r) {
            int m = r + h * 8;                          // C layout row
            smA[r] = sender[baseA + m];  rmA[r] = receiver[baseA + m];
            smB[r] = sender[baseB + m];  rmB[r] = receiver[baseB + m];
        }
        loadB(bfb[0], w2base, 0, lane);
        #pragma unroll
        for (int t = 0; t < 4; ++t) {
            v8f accA2, accB2;
            float bb = biasL[256 + t * 16 + nlo];
            #pragma unroll
            for (int j = 0; j < 8; ++j) { accA2[j] = bb; accB2[j] = bb; }
            #pragma unroll
            for (int c = 0; c < 4; ++c) {
                const int step = t * 4 + c;
                if (step < 15) {
                    const int tn = (c < 3) ? t : t + 1;
                    const int cn = (c < 3) ? c + 1 : 0;
                    loadB(bfb[(step + 1) & 1], w2base, cn * 4 + tn, lane);
                }
                accA2 = __builtin_amdgcn_wmma_f32_16x16x32_bf16(
                            false, a2A[c].v, false, bfb[step & 1].v, (short)0, accA2, false, false);
                accB2 = __builtin_amdgcn_wmma_f32_16x16x32_bf16(
                            false, a2B[c].v, false, bfb[step & 1].v, (short)0, accB2, false, false);
            }
            const int n = t * 16 + nlo;
            #pragma unroll
            for (int r = 0; r < 8; ++r) {
                float psiA = fmaxf(accA2[r], 0.0f);
                float dRA  = h_d[rmA[r] * DD + n];
                float dSA  = h_d[smA[r] * DD + n];
                unsafeAtomicAdd(&agg[rmA[r] * DD + n], psiA * (dRA - dSA));
                float psiB = fmaxf(accB2[r], 0.0f);
                float dRB  = h_d[rmB[r] * DD + n];
                float dSB  = h_d[smB[r] * DD + n];
                unsafeAtomicAdd(&agg[rmB[r] * DD + n], psiB * (dRB - dSB));
            }
        }
    }
}

// out = h_d_prev + agg @ Wm   (Wm staged as LDS B-fragments, WMMA per 16-row tile)
__global__ void __launch_bounds__(256)
out_kernel(const float* __restrict__ agg, const float* __restrict__ h_d,
           const float* __restrict__ Wm, float* __restrict__ out, int numTiles)
{
    __shared__ unsigned short wm[2 * 4 * 32 * 16];   // 8 KB
    const int tid = threadIdx.x;
    for (int idx = tid; idx < 4096; idx += 256) {
        int k = idx >> 6, n = idx & 63;
        int c = k >> 5, kk = k & 31;
        int hh = (kk >> 3) & 1;
        int ii = ((kk & 16) >> 1) | (kk & 7);
        int t  = n >> 4;
        int ln = (n & 15) + (hh << 4);
        wm[(((c * 4 + t) * 32 + ln) << 4) + ii] = __builtin_bit_cast(unsigned short, (__bf16)Wm[idx]);
    }
    __syncthreads();

    const int lane = tid & 31, wave = tid >> 5;
    const int nlo = lane & 15, h = lane >> 4;

    for (long tile = (long)blockIdx.x * 8 + wave; tile < numTiles; tile += (long)gridDim.x * 8) {
        const int base = (int)tile * 16;
        BF16Frag a[2];
        #pragma unroll
        for (int c = 0; c < 2; ++c) {
            F4x4 f = loadChunk(agg, (base + nlo) * DD + c * 32 + h * 8);
            cvtFrag(a[c], f);
        }
        #pragma unroll
        for (int t = 0; t < 4; ++t) {
            v8f acc;
            #pragma unroll
            for (int j = 0; j < 8; ++j) acc[j] = 0.0f;
            #pragma unroll
            for (int c = 0; c < 2; ++c) {
                BF16Frag bf;
                loadB(bf, wm, c * 4 + t, lane);
                acc = __builtin_amdgcn_wmma_f32_16x16x32_bf16(
                          false, a[c].v, false, bf.v, (short)0, acc, false, false);
            }
            const int n = t * 16 + nlo;
            #pragma unroll
            for (int r = 0; r < 8; ++r) {
                int o = (base + r + h * 8) * DD + n;
                out[o] = h_d[o] + acc[r];
            }
        }
    }
}

extern "C" void kernel_launch(void* const* d_in, const int* in_sizes, int n_in,
                              void* d_out, int out_size, void* d_ws, size_t ws_size,
                              hipStream_t stream) {
    const float* h_d_prev  = (const float*)d_in[0];
    const float* h_s       = (const float*)d_in[1];
    const float* edge_feat = (const float*)d_in[2];
    const int*   sender    = (const int*)d_in[3];
    const int*   receiver  = (const int*)d_in[4];
    const float* W0 = (const float*)d_in[5];
    const float* b0 = (const float*)d_in[6];
    const float* W1 = (const float*)d_in[7];
    const float* b1 = (const float*)d_in[8];
    const float* W2 = (const float*)d_in[9];
    const float* b2 = (const float*)d_in[10];
    const float* Wm = (const float*)d_in[11];
    float* out = (float*)d_out;

    const int N = in_sizes[0] / DD;   // 50000
    const int E = in_sizes[3];        // 800000
    float* agg = (float*)d_ws;
    const long aggN = (long)N * DD;

    zero_kernel<<<2048, 256, 0, stream>>>(agg, aggN);

    // 128KB weight frags + 64KB staging + biases
    const int ldsBytes = 198656;
    hipFuncSetAttribute(reinterpret_cast<const void*>(&edge_mlp_kernel),
                        hipFuncAttributeMaxDynamicSharedMemorySize, ldsBytes);

    const int numPairs = E / 32;      // 25000 pairs of 16-edge tiles
    edge_mlp_kernel<<<1024, 256, ldsBytes, stream>>>(
        h_d_prev, h_s, edge_feat, sender, receiver,
        W0, b0, W1, b1, W2, b2, agg, numPairs);

    const int nodeTiles = N / 16;     // 3125
    out_kernel<<<(nodeTiles + 7) / 8, 256, 0, stream>>>(agg, h_d_prev, Wm, out, nodeTiles);
}
